// NonLocalAttention_66846870995635
// MI455X (gfx1250) — matrix operand order repaired and checked by reading, not compiled, not run on hardware
//
#include <hip/hip_runtime.h>
#include <hip/hip_bf16.h>

// ---------------------------------------------------------------------------
// NonLocalAttention on MI455X (gfx1250): flash-attention with bf16 WMMA and
// double-buffered async global->LDS staging of K/V tiles.
// Shapes: N=4, C=256, CI=128, H=W=64 -> HW=4096.
// ---------------------------------------------------------------------------

constexpr int NB  = 4;
constexpr int CH  = 256;   // C
constexpr int CIH = 128;   // inter channels
constexpr int HWL = 4096;  // H*W

typedef __attribute__((ext_vector_type(16))) __bf16        bf16x16;
typedef __attribute__((ext_vector_type(8)))  float         f32x8;
typedef __attribute__((ext_vector_type(4)))  unsigned int  u32x4;
typedef __attribute__((ext_vector_type(4)))  int           i32x4;

// AS-qualified int4 pointers: the async-LDS builtin takes (v4i AS1*, v4i AS3*)
typedef __attribute__((address_space(1))) i32x4* g_i32x4_p;
typedef __attribute__((address_space(3))) i32x4* l_i32x4_p;

union Frag {
    u32x4   u[2];   // two 16-byte halves
    bf16x16 v;      // 16 bf16 elements (one WMMA 16-bit operand per lane)
};

#if defined(__has_builtin)
#if __has_builtin(__builtin_amdgcn_global_load_async_to_lds_b128)
#define HAVE_ASYNC_LDS 1
#endif
#endif

__device__ __forceinline__ unsigned short f2bf(float f) {
    unsigned u = __float_as_uint(f);
    unsigned r = (u + 0x7FFFu + ((u >> 16) & 1u)) >> 16;   // round-nearest-even
    return (unsigned short)r;
}

__device__ __forceinline__ f32x8 wmma_bf16(bf16x16 a, bf16x16 b, f32x8 c) {
    // D = A(16x32) * B(32x16) + C(16x16), f32 accumulate
    return __builtin_amdgcn_wmma_f32_16x16x32_bf16(false, a, false, b,
                                                   (short)0, c, false, false);
}

// 16-byte global -> LDS copy (async on gfx1250 toolchains that expose it)
__device__ __forceinline__ void copy16_g2l(const unsigned short* g,
                                           unsigned short* l) {
#ifdef HAVE_ASYNC_LDS
    __builtin_amdgcn_global_load_async_to_lds_b128(
        (g_i32x4_p)(uintptr_t)g, (l_i32x4_p)(unsigned)(uintptr_t)l, 0, 0);
#else
    *(u32x4*)l = *(const u32x4*)g;
#endif
}

__device__ __forceinline__ void wait_async0(void) {
#ifdef HAVE_ASYNC_LDS
#if __has_builtin(__builtin_amdgcn_s_wait_asynccnt)
    __builtin_amdgcn_s_wait_asynccnt(0);
#else
    asm volatile("s_wait_asynccnt 0" ::: "memory");
#endif
#endif
}

// one staged chunk = 4 async ops per thread still in flight
__device__ __forceinline__ void wait_async_le4(void) {
#ifdef HAVE_ASYNC_LDS
#if __has_builtin(__builtin_amdgcn_s_wait_asynccnt)
    __builtin_amdgcn_s_wait_asynccnt(4);
#else
    asm volatile("s_wait_asynccnt 4" ::: "memory");
#endif
#endif
}

// ---------------------------------------------------------------------------
// Kernel 1: the three 1x1-conv projections, written in WMMA-friendly bf16
// layouts.  grid = (HW/256, CI, N*3), block = 256.
//   which==0: theta(x) -> Q [n][pos][ci]
//   which==1: phi(y)   -> K [n][pos][ci]
//   which==2: g(x)     -> V [n][ci][pos]
// ---------------------------------------------------------------------------
__global__ __launch_bounds__(256) void proj_kernel(
    const float* __restrict__ x, const float* __restrict__ y,
    const float* __restrict__ tw, const float* __restrict__ tb,
    const float* __restrict__ pw, const float* __restrict__ pb,
    const float* __restrict__ gw, const float* __restrict__ gb,
    unsigned short* __restrict__ Q, unsigned short* __restrict__ K,
    unsigned short* __restrict__ V)
{
    const int pos   = blockIdx.x * 256 + threadIdx.x;
    const int ci    = blockIdx.y;
    const int nz    = blockIdx.z;
    const int n     = nz / 3;
    const int which = nz % 3;

    const float* src = (which == 1) ? y : x;
    const float* w   = (which == 0) ? tw : (which == 1) ? pw : gw;
    const float* b   = (which == 0) ? tb : (which == 1) ? pb : gb;

    const float* s  = src + (size_t)n * CH * HWL + pos;
    const float* wr = w + (size_t)ci * CH;

    float acc = b[ci];
#pragma unroll 4
    for (int c = 0; c < CH; ++c)
        acc = fmaf(wr[c], s[(size_t)c * HWL], acc);

    const unsigned short h = f2bf(acc);
    if (which == 0)      Q[((size_t)n * HWL + pos) * CIH + ci] = h;
    else if (which == 1) K[((size_t)n * HWL + pos) * CIH + ci] = h;
    else                 V[((size_t)n * CIH + ci) * HWL + pos] = h;
}

// ---------------------------------------------------------------------------
// Kernel 2: flash attention.  8 waves/block, 1 wave = one 16-query tile.
// Per 32-key chunk the block cooperatively stages the K tile (32x128 bf16,
// fully contiguous) and V tile (128x32 bf16) into LDS with double buffering
// via async global->LDS copies; each staged tile feeds 8 waves (128 queries).
// All K fragments are pulled into registers before the QK wmma chain; all V
// fragments are issued before the softmax so ds latency hides under v_exp.
// grid = (HW/128, N), block = 256.
// ---------------------------------------------------------------------------
__global__ __launch_bounds__(256) void attn_kernel(
    const unsigned short* __restrict__ Q,
    const unsigned short* __restrict__ K,
    const unsigned short* __restrict__ V,
    float* __restrict__ Oout)
{
    __shared__ __align__(16) unsigned short kTile[2][32 * 128];  // 2 x 8KB
    __shared__ __align__(16) unsigned short vTile[2][128 * 32];  // 2 x 8KB
    __shared__ __align__(16) unsigned short pLds[8][16 * 32];    // 8KB

    const int t    = threadIdx.x;
    const int lane = t & 31;
    const int wave = t >> 5;
    const int lh   = lane >> 4;     // lane half: 0 or 1
    const int ll   = lane & 15;
    const int n    = blockIdx.y;
    const int q0   = (blockIdx.x * 8 + wave) * 16;

    const unsigned short* Qn = Q + (size_t)n * HWL * CIH;
    const unsigned short* Kn = K + (size_t)n * HWL * CIH;
    const unsigned short* Vn = V + (size_t)n * CIH * HWL;

    // --- preload Q A-fragments (loop invariant), 4 slices of K-dim 32 ------
    // A layout: lane ll(+16*lh) holds row M=ll, K runs lh*16..lh*16+15
    Frag qf[4];
    {
        const unsigned short* qrow = Qn + (size_t)(q0 + ll) * CIH;
#pragma unroll
        for (int kc = 0; kc < 4; ++kc) {
            const int cb = kc * 32 + lh * 16;
            qf[kc].u[0] = *reinterpret_cast<const u32x4*>(qrow + cb);
            qf[kc].u[1] = *reinterpret_cast<const u32x4*>(qrow + cb + 8);
        }
    }

    float m[8], l[8];
    f32x8 Oacc[8];
#pragma unroll
    for (int r = 0; r < 8; ++r) { m[r] = -1e30f; l[r] = 0.0f; }
#pragma unroll
    for (int j = 0; j < 8; ++j) Oacc[j] = (f32x8){};

    unsigned short* pwv = &pLds[wave][0];

    // stage helper: 256 threads, 4 async b128 per thread per chunk.
    auto stage = [&](int key0, int buf) {
        // K tile: 32x128 bf16, fully contiguous 8KB; thread t copies 32B
        const unsigned short* ks = Kn + (size_t)key0 * CIH + t * 16;
        unsigned short*       kd = &kTile[buf][0] + t * 16;
        copy16_g2l(ks, kd);
        copy16_g2l(ks + 8, kd + 8);
        // V tile: 128 rows x 32 keys; thread t copies half of row t/2 (32B)
        const int vrow = t >> 1;
        const int vh   = (t & 1) * 16;
        const unsigned short* vs = Vn + (size_t)vrow * HWL + key0 + vh;
        unsigned short*       vd = &vTile[buf][0] + vrow * 32 + vh;
        copy16_g2l(vs, vd);
        copy16_g2l(vs + 8, vd + 8);
    };

    constexpr int NCHUNK = HWL / 32;   // 128
    stage(0, 0);                       // prologue: fill buffer 0

    for (int c = 0; c < NCHUNK; ++c) {
        const int buf = c & 1;
        if (c + 1 < NCHUNK) {
            stage((c + 1) * 32, buf ^ 1);   // issue next chunk's copies
            wait_async_le4();               // current chunk's copies done
        } else {
            wait_async0();
        }
        __syncthreads();                    // LDS tiles visible block-wide

        const unsigned short* kb = &kTile[buf][0];
        const unsigned short* vb = &vTile[buf][0];

        // --- pull all 8 K B-fragments into registers first -----------------
        // B layout: lane ll(+16*lh) holds column N=ll (a key), rows
        // K = lh*16 .. lh*16+15 -> 16 contiguous bf16 in [key][ci] tile.
        Frag kf0[4], kf1[4];
        {
            const unsigned short* kr0 = kb + ll * CIH + lh * 16;
            const unsigned short* kr1 = kr0 + 16 * CIH;
#pragma unroll
            for (int kc = 0; kc < 4; ++kc) {
                kf0[kc].u[0] = *reinterpret_cast<const u32x4*>(kr0 + kc * 32);
                kf0[kc].u[1] = *reinterpret_cast<const u32x4*>(kr0 + kc * 32 + 8);
                kf1[kc].u[0] = *reinterpret_cast<const u32x4*>(kr1 + kc * 32);
                kf1[kc].u[1] = *reinterpret_cast<const u32x4*>(kr1 + kc * 32 + 8);
            }
        }

        // --- S = Q . K^T : 8 back-to-back wmmas ----------------------------
        f32x8 S0 = (f32x8){};
        f32x8 S1 = (f32x8){};
#pragma unroll
        for (int kc = 0; kc < 4; ++kc) {
            S0 = wmma_bf16(qf[kc].v, kf0[kc].v, S0);
            S1 = wmma_bf16(qf[kc].v, kf1[kc].v, S1);
        }

        // --- issue all 8 V B-fragment loads; latency hides under softmax ---
        // B layout: lane ll(+16*lh) holds column N=ll (a ci), rows
        // K = keys lh*16 .. +15 -> contiguous in [ci][key] tile.
        Frag vf[8];
#pragma unroll
        for (int j = 0; j < 8; ++j) {
            const unsigned short* vp = vb + (j * 16 + ll) * 32 + lh * 16;
            vf[j].u[0] = *reinterpret_cast<const u32x4*>(vp);
            vf[j].u[1] = *reinterpret_cast<const u32x4*>(vp + 8);
        }

        // --- online softmax: C-layout row r lives in VGPR r per lane-half --
        float alpha[8];
#pragma unroll
        for (int r = 0; r < 8; ++r) {
            float v = fmaxf(S0[r], S1[r]);
#pragma unroll
            for (int off = 8; off; off >>= 1)
                v = fmaxf(v, __shfl_xor(v, off, 16));
            const float nm = fmaxf(m[r], v);
            const float a  = __expf(m[r] - nm);
            m[r] = nm;
            const float p0 = __expf(S0[r] - nm);
            const float p1 = __expf(S1[r] - nm);
            S0[r] = p0;
            S1[r] = p1;
            float rs = p0 + p1;
#pragma unroll
            for (int off = 8; off; off >>= 1)
                rs += __shfl_xor(rs, off, 16);
            l[r]     = l[r] * a + rs;
            alpha[r] = a;
        }
#pragma unroll
        for (int j = 0; j < 8; ++j)
#pragma unroll
            for (int r = 0; r < 8; ++r) Oacc[j][r] *= alpha[r];

        // --- P: C-layout -> A-fragment layout via LDS (per-wave tile) ------
#pragma unroll
        for (int r = 0; r < 8; ++r) {
            const int row = r + lh * 8;
            pwv[row * 32 + ll]      = f2bf(S0[r]);
            pwv[row * 32 + 16 + ll] = f2bf(S1[r]);
        }
        __asm__ volatile("s_wait_dscnt 0" ::: "memory");  // same-wave LDS RAW
        Frag pf;
        {
            const unsigned short* pr = pwv + ll * 32 + lh * 16;
            pf.u[0] = *reinterpret_cast<const u32x4*>(pr);
            pf.u[1] = *reinterpret_cast<const u32x4*>(pr + 8);
        }

        // --- O += P . V : 8 back-to-back wmmas -----------------------------
#pragma unroll
        for (int j = 0; j < 8; ++j)
            Oacc[j] = wmma_bf16(pf.v, vf[j].v, Oacc[j]);

        __syncthreads();   // all waves done with this buffer before reuse
    }

    // --- normalize by l and write O as [n][HW][CI] f32 ---------------------
#pragma unroll
    for (int r = 0; r < 8; ++r) l[r] = 1.0f / l[r];
#pragma unroll
    for (int j = 0; j < 8; ++j)
#pragma unroll
        for (int r = 0; r < 8; ++r) {
            const int q = q0 + r + lh * 8;
            Oout[((size_t)n * HWL + q) * CIH + j * 16 + ll] = Oacc[j][r] * l[r];
        }
}

// ---------------------------------------------------------------------------
// Kernel 3: rec 1x1 conv.  grid = (HW, N), block = 256 (= one thread per C).
// ---------------------------------------------------------------------------
__global__ __launch_bounds__(256) void rec_kernel(
    const float* __restrict__ attnO, const float* __restrict__ rec_w,
    const float* __restrict__ rec_b, float* __restrict__ recOut)
{
    __shared__ float arow[CIH];
    const int pos = blockIdx.x;
    const int n   = blockIdx.y;
    const int c   = threadIdx.x;

    const float* ar = attnO + ((size_t)n * HWL + pos) * CIH;
    if (c < CIH) arow[c] = ar[c];
    __syncthreads();

    const float* wr = rec_w + (size_t)c * CIH;
    float acc = rec_b[c];
#pragma unroll 4
    for (int k = 0; k < CIH; ++k) acc = fmaf(wr[k], arow[k], acc);
    recOut[((size_t)n * CH + c) * HWL + pos] = acc;
}

// ---------------------------------------------------------------------------
// Kernel 4: per-channel batch statistics.  grid = C, block = 256.
// ---------------------------------------------------------------------------
__global__ __launch_bounds__(256) void stats_kernel(
    const float* __restrict__ recOut, float* __restrict__ meanOut,
    float* __restrict__ rstdOut)
{
    __shared__ float s1[256];
    __shared__ float s2[256];
    const int c = blockIdx.x;
    const int t = threadIdx.x;

    float a = 0.0f, b = 0.0f;
    for (int n = 0; n < NB; ++n) {
        const float* p = recOut + ((size_t)n * CH + c) * HWL;
        for (int i = t; i < HWL; i += 256) {
            const float v = p[i];
            a += v;
            b += v * v;
        }
    }
    s1[t] = a; s2[t] = b;
    __syncthreads();
    for (int off = 128; off; off >>= 1) {
        if (t < off) { s1[t] += s1[t + off]; s2[t] += s2[t + off]; }
        __syncthreads();
    }
    if (t == 0) {
        const float inv  = 1.0f / (float)(NB * HWL);
        const float mean = s1[0] * inv;
        const float var  = s2[0] * inv - mean * mean;
        meanOut[c] = mean;
        rstdOut[c] = rsqrtf(var + 1e-5f);
    }
}

// ---------------------------------------------------------------------------
// Kernel 5: BN apply + residual.  grid = N*C*HW/256.
// ---------------------------------------------------------------------------
__global__ __launch_bounds__(256) void apply_kernel(
    const float* __restrict__ x, const float* __restrict__ recOut,
    const float* __restrict__ gamma, const float* __restrict__ beta,
    const float* __restrict__ mean, const float* __restrict__ rstd,
    float* __restrict__ out)
{
    const size_t i = (size_t)blockIdx.x * 256 + threadIdx.x;
    const int c = (int)((i / HWL) % CH);
    out[i] = x[i] + gamma[c] * (recOut[i] - mean[c]) * rstd[c] + beta[c];
}

// ---------------------------------------------------------------------------
extern "C" void kernel_launch(void* const* d_in, const int* in_sizes, int n_in,
                              void* d_out, int out_size, void* d_ws,
                              size_t ws_size, hipStream_t stream)
{
    const float* x   = (const float*)d_in[0];
    const float* y   = (const float*)d_in[1];
    const float* tw  = (const float*)d_in[2];
    const float* tb  = (const float*)d_in[3];
    const float* pw  = (const float*)d_in[4];
    const float* pb  = (const float*)d_in[5];
    const float* gw  = (const float*)d_in[6];
    const float* gb  = (const float*)d_in[7];
    const float* rw  = (const float*)d_in[8];
    const float* rb  = (const float*)d_in[9];
    const float* gam = (const float*)d_in[10];
    const float* bet = (const float*)d_in[11];
    float* out = (float*)d_out;

    // workspace carve (all sizes 256B-aligned by construction)
    char* ws = (char*)d_ws;
    const size_t bfMat = (size_t)NB * HWL * CIH * sizeof(unsigned short); // 4MB
    unsigned short* Q = (unsigned short*)(ws);
    unsigned short* K = (unsigned short*)(ws + bfMat);
    unsigned short* V = (unsigned short*)(ws + 2 * bfMat);
    float* attnO  = (float*)(ws + 3 * bfMat);                             // 8MB
    float* recOut = attnO + (size_t)NB * HWL * CIH;                       // 16MB
    float* meanB  = recOut + (size_t)NB * CH * HWL;
    float* rstdB  = meanB + CH;

    // 1) projections -> bf16 Q/K/V
    proj_kernel<<<dim3(HWL / 256, CIH, NB * 3), 256, 0, stream>>>(
        x, y, tw, tb, pw, pb, gw, gb, Q, K, V);

    // 2) flash attention (WMMA bf16, async-LDS double buffering)
    attn_kernel<<<dim3(HWL / 128, NB), 256, 0, stream>>>(Q, K, V, attnO);

    // 3) rec projection
    rec_kernel<<<dim3(HWL, NB), 256, 0, stream>>>(attnO, rw, rb, recOut);

    // 4) batch-norm statistics
    stats_kernel<<<dim3(CH), 256, 0, stream>>>(recOut, meanB, rstdB);

    // 5) normalize + residual
    apply_kernel<<<dim3((NB * CH * HWL) / 256), 256, 0, stream>>>(
        x, recOut, gam, bet, meanB, rstdB, out);
}
